// ESN_67499706024319
// MI455X (gfx1250) — matrix operand order repaired and checked by reading, not compiled, not run on hardware
//
#include <hip/hip_runtime.h>
#include <hip/hip_bf16.h>
#include <math.h>

// ---------------------------------------------------------------------------
// ESN forward on MI455X (gfx1250, wave32).
//  Kernel 1: esn_init  - zero barrier counter + state double-buffer.
//  Kernel 2: esn_proj  - U = input @ W_in^T + bias  (dense bf16 WMMA GEMM).
//  Kernel 3: esn_scan  - persistent 32-WG kernel; W_res (bf16) resident in
//            LDS (256KB/WGP of the 320KB CDNA5 LDS); per-step WMMA GEMV with
//            x in column 0 of B; global atomic barrier per step; state
//            re-broadcast via GLOBAL_LOAD_ASYNC_TO_LDS (ASYNCcnt).
// ---------------------------------------------------------------------------

typedef __attribute__((ext_vector_type(16))) __bf16 v16bf;
typedef __attribute__((ext_vector_type(8)))  float  v8f;

#define T_SEQ 4096
#define IN_SZ 128
#define R_SZ  2048
#define NWG   32
#define ROWS_PER_WG 64         // 2048 / 32
#define WAVES 8
#define BLOCK 256

// workspace layout (bytes)
#define WS_CTR_OFF 0
#define WS_XB_OFF  1024        // __bf16 xb[2][2048]  (double-buffered state)
#define WS_U_OFF   16384       // float  U[4096][2048] (input projection)

// LDS layout for scan kernel (bytes)
#define LDS_W_BYTES   262144   // 256 tiles * 1024B (bf16 A-fragments)
#define LDS_X_OFF     262144   // __bf16 x[2048]
#define LDS_S_OFF     266240   // float  s_own[64]
#define LDS_RED_OFF   266496   // float  red[4][2][8]
#define LDS_TOTAL     266752

// K index inside a 16x32 bf16 A-fragment for (vgpr-pair v, lane-half half)
// per CDNA5 ISA: v0..3 -> K=2v / 8+2v ; v4..7 -> 16+2(v-4) / 24+...
__device__ __forceinline__ int akbase(int v, int half) {
    return ((v < 4) ? (2 * v) : (16 + 2 * (v - 4))) + 8 * half;
}

// Async copy of 16 bytes global -> LDS for this lane (CDNA5 TDM-adjacent
// async path, tracked by ASYNCcnt). LDS byte address = low 32 bits of the
// generic LDS pointer; caller must s_wait_asynccnt + barrier before reading.
__device__ __forceinline__ void async_g2l_b128(void* lds_dst, const void* gsrc) {
    unsigned           ldsa = (unsigned)(unsigned long long)lds_dst;
    unsigned long long ga   = (unsigned long long)gsrc;
    asm volatile("global_load_async_to_lds_b128 %0, %1, off"
                 :: "v"(ldsa), "v"(ga) : "memory");
}
__device__ __forceinline__ void wait_async0() {
    asm volatile("s_wait_asynccnt 0x0" ::: "memory");
}

// ---------------------------------------------------------------------------
__global__ void esn_init(unsigned* __restrict__ ctr, __bf16* __restrict__ xb) {
    const int tid = threadIdx.x;
    if (tid == 0) *ctr = 0u;
    for (int i = tid; i < 2 * R_SZ; i += BLOCK) xb[i] = (__bf16)0.0f;
}

// ---------------------------------------------------------------------------
// U[t][r] = sum_k input[t][k] * W_in[r][k] + bias[r]
// One wave per 16x16 output tile; K = 128 -> 4 x wmma_f32_16x16x32_bf16.
__global__ void __launch_bounds__(BLOCK)
esn_proj(const float* __restrict__ inp, const float* __restrict__ Win,
         const float* __restrict__ bias, float* __restrict__ U) {
    const int wave = threadIdx.x >> 5;
    const int L    = threadIdx.x & 31;
    const int tile = blockIdx.x * WAVES + wave;     // 0 .. 32767
    const int t0   = (tile >> 7) * 16;              // M tile (time)
    const int n0   = (tile & 127) * 16;             // N tile (reservoir)
    const int m    = L & 15;
    const int half = L >> 4;

    v8f acc = {};
    for (int kt = 0; kt < 4; ++kt) {
        v16bf a, b;
        // A fragment: row m, K pairs per ISA layout
        #pragma unroll
        for (int v = 0; v < 8; ++v) {
            const int k = kt * 32 + akbase(v, half);
            const float* p = inp + (t0 + m) * IN_SZ + k;
            a[2 * v]     = (__bf16)p[0];
            a[2 * v + 1] = (__bf16)p[1];
        }
        // B fragment: column n = m, K pairs per ISA layout (same as A)
        #pragma unroll
        for (int j = 0; j < 8; ++j) {
            const int kB = kt * 32 + akbase(j, half);
            b[2 * j]     = (__bf16)Win[(n0 + m) * IN_SZ + kB];
            b[2 * j + 1] = (__bf16)Win[(n0 + m) * IN_SZ + kB + 1];
        }
        acc = __builtin_amdgcn_wmma_f32_16x16x32_bf16(
                  false, a, false, b, (short)0, acc, false, false);
    }
    // D layout: lane = column N; vgpr r -> row half*8 + r
    const int col = n0 + (L & 15);
    const float bv = bias[col];
    #pragma unroll
    for (int r = 0; r < 8; ++r) {
        const int row = t0 + half * 8 + r;
        U[row * R_SZ + col] = acc[r] + bv;
    }
}

// ---------------------------------------------------------------------------
__global__ void __launch_bounds__(BLOCK, 1)
esn_scan(const float* __restrict__ Wres, const float* __restrict__ U,
         float* __restrict__ out, __bf16* xb, unsigned* ctr) {
    extern __shared__ char smem[];
    __bf16* Wlds = (__bf16*)smem;                    // pre-swizzled A fragments
    __bf16* xlds = (__bf16*)(smem + LDS_X_OFF);      // staged state vector
    float*  sown = (float*) (smem + LDS_S_OFF);      // own fp32 state slice
    float*  red  = (float*) (smem + LDS_RED_OFF);    // cross-K reduction

    const int tid  = threadIdx.x;
    const int wave = tid >> 5;
    const int L    = tid & 31;
    const int r0   = blockIdx.x * ROWS_PER_WG;

    // ---- prologue: W_res rows -> bf16 A-fragment layout in LDS (once) ----
    // tile tIdx = rb*64 + kt covers rows [r0+rb*16, +16), K [kt*32, +32).
    for (int tIdx = wave; tIdx < 256; tIdx += WAVES) {
        const int rb = tIdx >> 6, kt = tIdx & 63;
        const int row  = r0 + rb * 16 + (L & 15);
        const int half = L >> 4;
        __bf16* dst = Wlds + tIdx * 512 + L * 16;    // 16 bf16 per lane, 32B
        #pragma unroll
        for (int v = 0; v < 8; ++v) {
            const int k = kt * 32 + akbase(v, half);
            dst[2 * v]     = (__bf16)Wres[row * R_SZ + k];
            dst[2 * v + 1] = (__bf16)Wres[row * R_SZ + k + 1];
        }
    }
    for (int i = tid; i < ROWS_PER_WG; i += BLOCK) sown[i] = 0.0f;

    // stage x for step 0 (phase 0, zeroed by esn_init): async global->LDS
    async_g2l_b128(xlds + tid * 8, xb + tid * 8);
    wait_async0();
    __syncthreads();

    const int rb = wave & 3;         // row block (16 rows)
    const int h  = wave >> 2;        // K half (0: K<1024, 1: K>=1024)
    const float inv_sqrt = 0.022097086912079608f;    // 1/sqrt(2048)

    for (int t = 0; t < T_SEQ; ++t) {
        // ---- y[16 rows] partial over 1024 K via 32 WMMAs (col 0 of B) ----
        v8f acc = {};
        const int kt0 = h * 32;
        #pragma unroll 4
        for (int kt = kt0; kt < kt0 + 32; ++kt) {
            v16bf a = *(const v16bf*)(Wlds + (rb * 64 + kt) * 512 + L * 16);
            v16bf b;
            #pragma unroll
            for (int v = 0; v < 8; ++v) {            // K pairs per ISA layout
                const int kx = kt * 32 + akbase(v, L >> 4);
                b[2 * v]     = xlds[kx];
                b[2 * v + 1] = xlds[kx + 1];
            }
            acc = __builtin_amdgcn_wmma_f32_16x16x32_bf16(
                      false, a, false, b, (short)0, acc, false, false);
        }
        // column 0 of D lives in lanes 0 (rows 0..7) and 16 (rows 8..15)
        if (h == 1 && (L & 15) == 0) {
            const int hi = L >> 4;
            #pragma unroll
            for (int r = 0; r < 8; ++r) red[(rb * 2 + hi) * 8 + r] = acc[r];
        }
        __syncthreads();
        if (h == 0 && (L & 15) == 0) {
            const int hi = L >> 4;
            __bf16* xnext = xb + ((t + 1) & 1) * R_SZ;
            #pragma unroll
            for (int r = 0; r < 8; ++r) {
                const int mloc = rb * 16 + hi * 8 + r;
                const int rowg = r0 + mloc;
                const float y    = acc[r] + red[(rb * 2 + hi) * 8 + r];
                const float pre  = y + U[t * R_SZ + rowg];
                const float snew = 0.1f * sown[mloc]
                                 + 0.9f * erff(pre) * inv_sqrt;
                out[t * R_SZ + rowg] = snew;
                xnext[rowg] = (__bf16)snew;
                sown[mloc]  = snew;
            }
        }
        if (t + 1 < T_SEQ) {
            // warm L2/L1 for next step's U row slice while we sit in the
            // barrier (global_prefetch_b8)
            __builtin_prefetch(U + (t + 1) * R_SZ + tid * 8, 0, 1);
            __threadfence();                          // publish xnext
            __syncthreads();
            if (tid == 0) {
                __hip_atomic_fetch_add(ctr, 1u, __ATOMIC_RELEASE,
                                       __HIP_MEMORY_SCOPE_AGENT);
                const unsigned target = (unsigned)NWG * (unsigned)(t + 1);
                while (__hip_atomic_load(ctr, __ATOMIC_ACQUIRE,
                                         __HIP_MEMORY_SCOPE_AGENT) < target)
                    __builtin_amdgcn_s_sleep(2);
            }
            __syncthreads();
            __builtin_amdgcn_fence(__ATOMIC_ACQUIRE, "agent");
            // restage full state for next step: async global->LDS (ASYNCcnt)
            async_g2l_b128(xlds + tid * 8,
                           xb + ((t + 1) & 1) * R_SZ + tid * 8);
            wait_async0();
            __syncthreads();
        }
    }
}

// ---------------------------------------------------------------------------
extern "C" void kernel_launch(void* const* d_in, const int* in_sizes, int n_in,
                              void* d_out, int out_size, void* d_ws, size_t ws_size,
                              hipStream_t stream) {
    const float* inp  = (const float*)d_in[0];   // (4096, 128)
    const float* Win  = (const float*)d_in[1];   // (2048, 128)
    const float* Wres = (const float*)d_in[2];   // (2048, 2048)
    const float* bias = (const float*)d_in[3];   // (2048,)
    float* out = (float*)d_out;                  // (4096, 2048) fp32

    unsigned* ctr = (unsigned*)((char*)d_ws + WS_CTR_OFF);
    __bf16*   xb  = (__bf16*)  ((char*)d_ws + WS_XB_OFF);
    float*    U   = (float*)   ((char*)d_ws + WS_U_OFF);

    (void)in_sizes; (void)n_in; (void)out_size; (void)ws_size;

    hipFuncSetAttribute((const void*)esn_scan,
                        hipFuncAttributeMaxDynamicSharedMemorySize, LDS_TOTAL);

    esn_init<<<1, BLOCK, 0, stream>>>(ctr, xb);
    esn_proj<<<(T_SEQ / 16) * (R_SZ / 16) / WAVES, BLOCK, 0, stream>>>(
        inp, Win, bias, U);
    esn_scan<<<NWG, BLOCK, LDS_TOTAL, stream>>>(Wres, U, out, xb, ctr);
}
